// StackedLstm3_68066641707054
// MI455X (gfx1250) — compile-verified, matrix operand order checked by reading, not run on hardware
//
#include <hip/hip_runtime.h>
#include <hip/hip_bf16.h>

#define VOCAB 2048
#define HID   1024
#define BATCH 64
#define SEQ   256
#define G4    4096

typedef __attribute__((ext_vector_type(16))) __bf16 v16bf;
typedef __attribute__((ext_vector_type(8)))  __bf16 v8bf;
typedef __attribute__((ext_vector_type(8)))  float  v8f;

__device__ __forceinline__ float sigf(float x)     { return 1.f / (1.f + __expf(-x)); }
__device__ __forceinline__ float tanhfastf(float x){ float e = __expf(-2.f * x); return (1.f - e) / (1.f + e); }

// WMMA with compile-time reuse-B hint (B fragment shared by consecutive calls)
template<bool RB>
__device__ __forceinline__ v8f wmma_bf16(v16bf a, v16bf b, v8f c) {
  return __builtin_amdgcn_wmma_f32_16x16x32_bf16(false, a, false, b, (short)0, c, false, RB);
}

// ---------------------------------------------------------------------------
// One-time: convert fp32 W[K][N] (row-major) -> bf16 W^T[N][K] so that WMMA
// B-operand fragments are contiguous 32B per lane.
// ---------------------------------------------------------------------------
__global__ void cvt_wT(const float* __restrict__ src, __bf16* __restrict__ dst,
                       int Kd, int Nd, int rowOff) {
  long id  = (long)blockIdx.x * blockDim.x + threadIdx.x;
  long tot = (long)Kd * Nd;
  if (id >= tot) return;
  int n = (int)(id / Kd);
  int k = (int)(id % Kd);
  dst[(long)n * Kd + k] = (__bf16)src[(long)(rowOff + k) * Nd + n];
}

__global__ void zero_b(float4* __restrict__ p, long n) {
  long i = (long)blockIdx.x * blockDim.x + threadIdx.x;
  if (i < n) p[i] = make_float4(0.f, 0.f, 0.f, 0.f);
}

// ---------------------------------------------------------------------------
// Fused LSTM cell step for one layer.
//   gates[64][4096] = concat(A0, A1)[64][KDIM] @ W^T  (+ bias, + W1[token] row for L1)
// Grid 64 WGs x 16 h-cols. 8 waves = 4 gates x 2 M-halves, 2 M-subtiles each.
// A tiles (64 x 64 bf16, 8 KB) staged with GLOBAL_LOAD_ASYNC_TO_LDS_B128,
// software-pipelined so stage(i+1) overlaps the WMMA consume of tile i.
// Full LSTM cell epilogue fused (gate exchange through LDS).
// ---------------------------------------------------------------------------
template<int KDIM, bool LAYER1>
__global__ __launch_bounds__(256)
void lstm_cell_k(const __bf16* __restrict__ WT,    // [4096][KDIM] bf16, transposed
                 const __bf16* __restrict__ A0,    // [64][1024] bf16
                 const __bf16* __restrict__ A1,    // [64][1024] bf16 or null
                 const float*  __restrict__ bias,  // [4096]
                 const float*  __restrict__ W1v,   // fp32 W1 base (vocab gather), L1 only
                 const int*    __restrict__ X,     // [64][256] tokens, L1 only
                 int t,
                 float*  __restrict__ Cst,         // [64][1024] fp32 running cell state
                 __bf16* __restrict__ Hw,          // [64][1024] bf16 h write buffer
                 float*  __restrict__ Hfin,        // fp32 final h (d_out)
                 float*  __restrict__ Cfin,        // fp32 final c (d_out)
                 int writeFinal)
{
  __shared__ __align__(64) __bf16 Ald[2][BATCH * 64];   // 16 KB double-buffered A tile
  __shared__ float  Gl[4][BATCH * 16];                  // 16 KB gate exchange

  const int tid  = threadIdx.x;
  const int wave = tid >> 5;
  const int lane = tid & 31;
  const int gate = wave & 3;
  const int mh   = wave >> 2;              // M half: rows [mh*32, mh*32+31]
  const int lr   = lane & 15;
  const int kh   = (lane >> 4) * 16;       // K sub-offset per half-wave
  const int colbase = gate * HID + blockIdx.x * 16;

  // async stage of one 64x64 bf16 A tile: 256 threads x 32 B, direct to LDS
  auto stage = [&](int it, int b) {
    const int kglob = it * 64;
    const __bf16* base = (kglob < HID) ? A0 : A1;
    const int kloc = (kglob < HID) ? kglob : (kglob - HID);
    const int row = tid >> 2;
    const int c16 = (tid & 3) << 4;
    unsigned long long ga =
        (unsigned long long)(uintptr_t)(base + row * HID + kloc + c16);
    unsigned la = (unsigned)(uintptr_t)&Ald[b][row * 64 + c16];
    asm volatile("global_load_async_to_lds_b128 %0, %1, off\n\t"
                 "global_load_async_to_lds_b128 %0, %1, off offset:16"
                 :: "v"(la), "v"(ga) : "memory");
  };

  v8f acc0 = v8f{}, acc1 = v8f{};

  constexpr int NIT = KDIM / 64;
  stage(0, 0);
  for (int it = 0; it < NIT; ++it) {
    asm volatile("s_wait_asynccnt 0x0" ::: "memory");   // tile `it` landed in LDS
    __syncthreads();
    if (it + 1 < NIT) stage(it + 1, (it + 1) & 1);      // overlaps compute below
    const int b = it & 1;

    #pragma unroll
    for (int ks = 0; ks < 2; ++ks) {                    // two 32-K WMMA slices
      const __bf16* bp = WT + (long)(colbase + lr) * KDIM + it * 64 + ks * 32 + kh;
      v16bf bF = *(const v16bf*)bp;
      __builtin_prefetch((const void*)(bp + 64), 0, 3); // next chunk, near-cache

      // preload both A fragments, then issue back-to-back WMMAs (B reused)
      v16bf aF0 = *(const v16bf*)&Ald[b][((mh * 2 + 0) * 16 + lr) * 64 + ks * 32 + kh];
      v16bf aF1 = *(const v16bf*)&Ald[b][((mh * 2 + 1) * 16 + lr) * 64 + ks * 32 + kh];
      acc0 = wmma_bf16<true >(aF0, bF, acc0);
      acc1 = wmma_bf16<false>(aF1, bF, acc1);
    }
  }

  // exchange gate pre-activations via LDS
  #pragma unroll
  for (int mt = 0; mt < 2; ++mt) {
    const v8f& a = mt ? acc1 : acc0;
    #pragma unroll
    for (int r = 0; r < 8; ++r) {
      const int m = (mh * 2 + mt) * 16 + ((lane >> 4) << 3) + r;
      Gl[gate][m * 16 + lr] = a[r];
    }
  }
  __syncthreads();

  // full cell epilogue: 256 threads x 4 elements = 64 x 16
  const int col  = tid & 15;
  const int r0   = (tid >> 4) * 4;
  const int gcol = blockIdx.x * 16 + col;
  #pragma unroll
  for (int j = 0; j < 4; ++j) {
    const int row = r0 + j;
    float f  = Gl[0][row * 16 + col] + bias[0 * HID + gcol];
    float i  = Gl[1][row * 16 + col] + bias[1 * HID + gcol];
    float o  = Gl[2][row * 16 + col] + bias[2 * HID + gcol];
    float ct = Gl[3][row * 16 + col] + bias[3 * HID + gcol];
    if (LAYER1) {
      const int tok = X[row * SEQ + t];            // one-hot @ W1 == row gather
      const float* wr = W1v + (long)tok * G4;
      f  += wr[0 * HID + gcol];
      i  += wr[1 * HID + gcol];
      o  += wr[2 * HID + gcol];
      ct += wr[3 * HID + gcol];
    }
    const float cprev = Cst[row * HID + gcol];
    const float cnew  = sigf(f) * cprev + sigf(i) * tanhfastf(ct);
    const float hnew  = sigf(o) * tanhfastf(cnew);
    Cst[row * HID + gcol] = cnew;
    Hw[row * HID + gcol]  = (__bf16)hnew;
    if (writeFinal) {
      Hfin[row * HID + gcol] = hnew;
      Cfin[row * HID + gcol] = cnew;
    }
  }
}

// ---------------------------------------------------------------------------
// Output projection: out[t*64 .. t*64+63][2048] = h3 @ Wout + bout   (fp32)
// Grid 32 WGs x 64 cols. 8 waves = 4 n-tiles x 2 M-halves, 2 M-subtiles each.
// ---------------------------------------------------------------------------
__global__ __launch_bounds__(256)
void out_proj_k(const __bf16* __restrict__ WoT,   // [2048][1024] bf16 transposed
                const __bf16* __restrict__ H3,    // [64][1024] bf16
                const float*  __restrict__ bout,  // [2048]
                float* __restrict__ out, int t)
{
  __shared__ __align__(64) __bf16 Ald[2][BATCH * 64];
  const int tid  = threadIdx.x;
  const int wave = tid >> 5;
  const int lane = tid & 31;
  const int nt   = wave & 3;               // n-tile (16 cols)
  const int mh   = wave >> 2;              // M half
  const int lr   = lane & 15;
  const int kh   = (lane >> 4) * 16;
  const int colbase = blockIdx.x * 64 + nt * 16;

  auto stage = [&](int it, int b) {
    const int row = tid >> 2;
    const int c16 = (tid & 3) << 4;
    unsigned long long ga =
        (unsigned long long)(uintptr_t)(H3 + row * HID + it * 64 + c16);
    unsigned la = (unsigned)(uintptr_t)&Ald[b][row * 64 + c16];
    asm volatile("global_load_async_to_lds_b128 %0, %1, off\n\t"
                 "global_load_async_to_lds_b128 %0, %1, off offset:16"
                 :: "v"(la), "v"(ga) : "memory");
  };

  v8f acc0 = v8f{}, acc1 = v8f{};
  constexpr int NIT = HID / 64;
  stage(0, 0);
  for (int it = 0; it < NIT; ++it) {
    asm volatile("s_wait_asynccnt 0x0" ::: "memory");
    __syncthreads();
    if (it + 1 < NIT) stage(it + 1, (it + 1) & 1);
    const int b = it & 1;

    #pragma unroll
    for (int ks = 0; ks < 2; ++ks) {
      const __bf16* bp = WoT + (long)(colbase + lr) * HID + it * 64 + ks * 32 + kh;
      v16bf bF = *(const v16bf*)bp;
      __builtin_prefetch((const void*)(bp + 64), 0, 3);

      v16bf aF0 = *(const v16bf*)&Ald[b][((mh * 2 + 0) * 16 + lr) * 64 + ks * 32 + kh];
      v16bf aF1 = *(const v16bf*)&Ald[b][((mh * 2 + 1) * 16 + lr) * 64 + ks * 32 + kh];
      acc0 = wmma_bf16<true >(aF0, bF, acc0);
      acc1 = wmma_bf16<false>(aF1, bF, acc1);
    }
  }

  const int gcol = colbase + lr;
  const float bb = bout[gcol];
  #pragma unroll
  for (int mt = 0; mt < 2; ++mt) {
    const v8f& a = mt ? acc1 : acc0;
    #pragma unroll
    for (int r = 0; r < 8; ++r) {
      const int m = (mh * 2 + mt) * 16 + ((lane >> 4) << 3) + r;
      out[(long)(t * BATCH + m) * VOCAB + gcol] = a[r] + bb;
    }
  }
}

// ---------------------------------------------------------------------------
extern "C" void kernel_launch(void* const* d_in, const int* in_sizes, int n_in,
                              void* d_out, int out_size, void* d_ws, size_t ws_size,
                              hipStream_t stream)
{
  const int*   X    = (const int*)  d_in[0];
  const float* W1   = (const float*)d_in[1];
  const float* b1   = (const float*)d_in[2];
  const float* W2   = (const float*)d_in[3];
  const float* b2   = (const float*)d_in[4];
  const float* W3   = (const float*)d_in[5];
  const float* b3   = (const float*)d_in[6];
  const float* Wout = (const float*)d_in[7];
  const float* bout = (const float*)d_in[8];

  float* out = (float*)d_out;                       // [16384][2048]
  float* H1f = out + (long)SEQ * BATCH * VOCAB;
  float* C1f = H1f + BATCH * HID;
  float* H2f = C1f + BATCH * HID;
  float* C2f = H2f + BATCH * HID;
  float* H3f = C2f + BATCH * HID;
  float* C3f = H3f + BATCH * HID;

  char* ws = (char*)d_ws;
  __bf16* w1T = (__bf16*)(ws);                      // [4096][1024]  8 MB
  __bf16* w2T = (__bf16*)(ws + (8l  << 20));        // [4096][2048] 16 MB
  __bf16* w3T = (__bf16*)(ws + (24l << 20));        // [4096][2048] 16 MB
  __bf16* woT = (__bf16*)(ws + (40l << 20));        // [2048][1024]  4 MB
  __bf16* h1b = (__bf16*)(ws + (44l << 20));        // [2][64][1024]
  __bf16* h2b = h1b + 2 * BATCH * HID;
  __bf16* h3b = h2b + 2 * BATCH * HID;
  float*  c1  = (float*)(h3b + 2 * BATCH * HID);
  float*  c2  = c1 + BATCH * HID;
  float*  c3  = c2 + BATCH * HID;

  // weight convert + transpose (fp32 -> bf16)
  cvt_wT<<<(HID   * G4)    / 256, 256, 0, stream>>>(W1,   w1T, HID,     G4,    VOCAB);
  cvt_wT<<<(2*HID * G4)    / 256, 256, 0, stream>>>(W2,   w2T, 2 * HID, G4,    0);
  cvt_wT<<<(2*HID * G4)    / 256, 256, 0, stream>>>(W3,   w3T, 2 * HID, G4,    0);
  cvt_wT<<<(HID   * VOCAB) / 256, 256, 0, stream>>>(Wout, woT, HID,     VOCAB, 0);

  // zero h (both buffers, bf16) and c (fp32): contiguous region from h1b
  const long zbytes = 3l * 2 * BATCH * HID * sizeof(__bf16)
                    + 3l * BATCH * HID * sizeof(float);
  const long z16 = zbytes / 16;
  zero_b<<<(int)((z16 + 255) / 256), 256, 0, stream>>>((float4*)h1b, z16);

  for (int t = 0; t < SEQ; ++t) {
    const int rd  = t & 1;
    const int wr  = rd ^ 1;
    const int fin = (t == SEQ - 1);
    lstm_cell_k<HID, true><<<64, 256, 0, stream>>>(
        w1T, h1b + rd * BATCH * HID, (const __bf16*)nullptr,
        b1, W1, X, t, c1, h1b + wr * BATCH * HID, H1f, C1f, fin);
    lstm_cell_k<2 * HID, false><<<64, 256, 0, stream>>>(
        w2T, h1b + wr * BATCH * HID, h2b + rd * BATCH * HID,
        b2, (const float*)nullptr, (const int*)nullptr, t,
        c2, h2b + wr * BATCH * HID, H2f, C2f, fin);
    lstm_cell_k<2 * HID, false><<<64, 256, 0, stream>>>(
        w3T, h2b + wr * BATCH * HID, h3b + rd * BATCH * HID,
        b3, (const float*)nullptr, (const int*)nullptr, t,
        c3, h3b + wr * BATCH * HID, H3f, C3f, fin);
    out_proj_k<<<32, 256, 0, stream>>>(
        woT, h3b + wr * BATCH * HID, bout, out, t);
  }
}
